// MimiAttention_28604482192014
// MI455X (gfx1250) — compile-verified, matrix operand order checked by reading, not run on hardware
//
#include <hip/hip_runtime.h>
#include <hip/hip_bf16.h>
#include <math.h>

typedef _Float16 f16;
typedef __attribute__((ext_vector_type(8)))  _Float16 v8h;
typedef __attribute__((ext_vector_type(16))) _Float16 v16h;
typedef __attribute__((ext_vector_type(8)))  float    v8f;

#define B_      2
#define S_      2048
#define D_      1024
#define H_      16
#define HK_     4
#define HD_     64
#define NREP_   (H_ / HK_)
#define WINDOW_ 250
#define SCALE_  0.125f
#define M_      (B_ * S_)      // 4096 rows for all projections

// ---------------------------------------------------------------------------
// CDNA5 async global->LDS copy (ASYNCcnt-tracked), ISA §10.7/§15.18.3.
// VDST = LDS byte address (low 32 bits of a generic LDS pointer),
// VADDR = 64-bit global address, GV mode.
// ---------------------------------------------------------------------------
__device__ __forceinline__ void async_g2l_b128(void* lds_ptr, const void* gptr) {
    unsigned int laddr = (unsigned int)(uintptr_t)lds_ptr;
    unsigned long long gaddr = (unsigned long long)(uintptr_t)gptr;
    asm volatile("global_load_async_to_lds_b128 %0, %1, off"
                 :: "v"(laddr), "v"(gaddr) : "memory");
}

__device__ __forceinline__ void wait_async0() {
#if __has_builtin(__builtin_amdgcn_s_wait_asynccnt)
    __builtin_amdgcn_s_wait_asynccnt(0);
#else
    asm volatile("s_wait_asynccnt 0x0" ::: "memory");
#endif
}

// ---------------------------------------------------------------------------
// Fragment loader for V_WMMA_F32_16X16X32_F16.
// 16-bit A-matrix 16x32 layout (ISA 7.12.2): lane m=lane&15, hi=lane>>4;
// VGPR v half h holds K = (v>=4 ? 16 : 0) + (v&3)*2 + h + hi*8.
// => per lane: 8 contiguous halves at k0+hi*8 and 8 at k0+16+hi*8 (two b128s).
// B fragment (lane n = row n of B^T) uses the identical packing.
// ---------------------------------------------------------------------------
__device__ __forceinline__ v16h load_frag(const f16* __restrict__ base, int ld, int k0) {
    int lane = threadIdx.x & 31;
    int m = lane & 15, hi = lane >> 4;
    const f16* p = base + (size_t)m * ld + k0 + hi * 8;
    v8h lo = *(const v8h*)(p);
    v8h hiq = *(const v8h*)(p + 16);
    return __builtin_shufflevector(lo, hiq, 0,1,2,3,4,5,6,7,8,9,10,11,12,13,14,15);
}

__device__ __forceinline__ v16h load_frag_lds(const f16* base, int ld, int k0) {
    int lane = threadIdx.x & 31;
    int m = lane & 15, hi = lane >> 4;
    const f16* p = base + m * ld + k0 + hi * 8;
    v8h lo = *(const v8h*)(p);
    v8h hiq = *(const v8h*)(p + 16);
    return __builtin_shufflevector(lo, hiq, 0,1,2,3,4,5,6,7,8,9,10,11,12,13,14,15);
}

// ---------------------------------------------------------------------------
// fp32 -> fp16 convert (grid-stride)
// ---------------------------------------------------------------------------
__global__ void k_cvt(const float* __restrict__ in, f16* __restrict__ out, int n) {
    int i = blockIdx.x * blockDim.x + threadIdx.x;
    int st = gridDim.x * blockDim.x;
    for (; i < n; i += st) out[i] = (f16)in[i];
}

// ---------------------------------------------------------------------------
// Y[M,N] = X[M,K] @ W[N,K]^T
// Block = 8 waves sharing one nj (N-band). The 64x32 W tile for each k-step is
// DMA'd into LDS once per block via global_load_async_to_lds_b128 (one b128
// per thread), double-buffered against compute. A-fragments stay on a
// register double-buffered direct-global path (no cross-wave A reuse).
// Each wave computes a 32x64 tile (8 WMMA per k-step).
// ---------------------------------------------------------------------------
template <bool F16OUT>
__global__ __launch_bounds__(256) void k_gemm(const f16* __restrict__ X,
                                              const f16* __restrict__ W,
                                              void* __restrict__ Yv,
                                              int M, int N, int K) {
    int tid = threadIdx.x;
    int wid = (int)((blockIdx.x * blockDim.x + tid) >> 5);
    int mtiles = M >> 5, ntiles = N >> 6;
    if (wid >= mtiles * ntiles) return;
    int mi = wid % mtiles, nj = wid / mtiles;   // mtiles%8==0 -> whole block shares nj
    int lane = tid & 31;
    int m = lane & 15, hi = lane >> 4;

    __shared__ __align__(16) f16 sW[2][64 * 32];   // double-buffered W tile (2x4KB)

    const f16* xr0 = X + (size_t)(mi * 32) * K;
    const f16* xr1 = xr0 + (size_t)16 * K;
    const f16* wr  = W + (size_t)(nj * 64) * K;

    // thread tid stages 16B of the tile: row = tid/4, 8-half segment = tid%4
    int srow = tid >> 2, sseg = (tid & 3) * 8;

    v8f acc[2][4] = {};

    // prologue: stage tile 0, preload A fragments for k0=0
    async_g2l_b128(&sW[0][srow * 32 + sseg], wr + (size_t)srow * K + sseg);
    v16h a0 = load_frag(xr0, K, 0);
    v16h a1 = load_frag(xr1, K, 0);
    wait_async0();
    __syncthreads();

    int cur = 0;
    for (int k0 = 0; k0 < K; k0 += 32) {
        int kn = k0 + 32;
        if (kn < K)   // stage next W tile into the other buffer (overlaps compute)
            async_g2l_b128(&sW[cur ^ 1][srow * 32 + sseg], wr + (size_t)srow * K + kn + sseg);

        int ka = (kn < K) ? kn : k0;               // next A fragments (benign reload at end)
        v16h na0 = load_frag(xr0, K, ka);
        v16h na1 = load_frag(xr1, K, ka);

        int kp = k0 + 64;                          // prefetch A two steps ahead
        if (kp < K) {
            __builtin_prefetch(xr0 + (size_t)m * K + kp + hi * 8, 0, 3);
            __builtin_prefetch(xr1 + (size_t)m * K + kp + hi * 8, 0, 3);
        }

        v16h b0 = load_frag_lds(&sW[cur][0 * 16 * 32], 32, 0);
        v16h b1 = load_frag_lds(&sW[cur][1 * 16 * 32], 32, 0);
        v16h b2 = load_frag_lds(&sW[cur][2 * 16 * 32], 32, 0);
        v16h b3 = load_frag_lds(&sW[cur][3 * 16 * 32], 32, 0);

        acc[0][0] = __builtin_amdgcn_wmma_f32_16x16x32_f16(false, a0, false, b0, (short)0, acc[0][0], false, false);
        acc[0][1] = __builtin_amdgcn_wmma_f32_16x16x32_f16(false, a0, false, b1, (short)0, acc[0][1], false, false);
        acc[0][2] = __builtin_amdgcn_wmma_f32_16x16x32_f16(false, a0, false, b2, (short)0, acc[0][2], false, false);
        acc[0][3] = __builtin_amdgcn_wmma_f32_16x16x32_f16(false, a0, false, b3, (short)0, acc[0][3], false, false);
        acc[1][0] = __builtin_amdgcn_wmma_f32_16x16x32_f16(false, a1, false, b0, (short)0, acc[1][0], false, false);
        acc[1][1] = __builtin_amdgcn_wmma_f32_16x16x32_f16(false, a1, false, b1, (short)0, acc[1][1], false, false);
        acc[1][2] = __builtin_amdgcn_wmma_f32_16x16x32_f16(false, a1, false, b2, (short)0, acc[1][2], false, false);
        acc[1][3] = __builtin_amdgcn_wmma_f32_16x16x32_f16(false, a1, false, b3, (short)0, acc[1][3], false, false);

        a0 = na0; a1 = na1;
        wait_async0();      // this wave's async writes to the other buffer done
        __syncthreads();    // all waves' writes done -> safe to read next tile
        cur ^= 1;
    }

#pragma unroll
    for (int mt = 0; mt < 2; ++mt) {
#pragma unroll
        for (int t = 0; t < 4; ++t) {
#pragma unroll
            for (int r = 0; r < 8; ++r) {
                size_t row = (size_t)(mi * 32 + mt * 16 + r + 8 * hi);
                size_t col = (size_t)(nj * 64 + t * 16 + m);
                float v = acc[mt][t][r];
                if (F16OUT) ((f16*)Yv)[row * N + col] = (f16)v;
                else        ((float*)Yv)[row * N + col] = v;
            }
        }
    }
}

// ---------------------------------------------------------------------------
// RoPE on Q (folds in SCALE) : qpre[B*S, H*64] -> q[B,H,S,64]
// ---------------------------------------------------------------------------
__global__ void k_rope_q(const f16* __restrict__ qp, f16* __restrict__ q) {
    int t = blockIdx.x * blockDim.x + threadIdx.x;
    if (t >= B_ * S_ * H_) return;
    int h = t % H_; int bs = t / H_; int s = bs % S_; int b = bs / S_;
    const f16* in = qp + (size_t)bs * (H_ * HD_) + h * HD_;
    f16* out = q + ((size_t)(b * H_ + h) * S_ + s) * HD_;
    float pos = (float)s;
    const float lg = logf(10000.0f) / 32.0f;
#pragma unroll
    for (int d = 0; d < 32; ++d) {
        float ang = pos * __expf(-lg * (float)d);
        float c = __cosf(ang), sn = __sinf(ang);
        float x1 = (float)in[d], x2 = (float)in[d + 32];
        out[d]      = (f16)((x1 * c - x2 * sn) * SCALE_);
        out[d + 32] = (f16)((x2 * c + x1 * sn) * SCALE_);
    }
}

// RoPE on K : kpre[B*S, HK*64] -> k[B,HK,S,64]
__global__ void k_rope_k(const f16* __restrict__ kp, f16* __restrict__ k) {
    int t = blockIdx.x * blockDim.x + threadIdx.x;
    if (t >= B_ * S_ * HK_) return;
    int hk = t % HK_; int bs = t / HK_; int s = bs % S_; int b = bs / S_;
    const f16* in = kp + (size_t)bs * (HK_ * HD_) + hk * HD_;
    f16* out = k + ((size_t)(b * HK_ + hk) * S_ + s) * HD_;
    float pos = (float)s;
    const float lg = logf(10000.0f) / 32.0f;
#pragma unroll
    for (int d = 0; d < 32; ++d) {
        float ang = pos * __expf(-lg * (float)d);
        float c = __cosf(ang), sn = __sinf(ang);
        float x1 = (float)in[d], x2 = (float)in[d + 32];
        out[d]      = (f16)(x1 * c - x2 * sn);
        out[d + 32] = (f16)(x2 * c + x1 * sn);
    }
}

// V transpose: vpre[B*S, HK*64] -> vt[B,HK,64,S]
__global__ void k_pack_v(const f16* __restrict__ vp, f16* __restrict__ vt) {
    int t = blockIdx.x * blockDim.x + threadIdx.x;
    if (t >= B_ * S_ * HK_) return;
    int hk = t % HK_; int bs = t / HK_; int s = bs % S_; int b = bs / S_;
    const f16* in = vp + (size_t)bs * (HK_ * HD_) + hk * HD_;
    f16* out = vt + ((size_t)(b * HK_ + hk) * HD_) * S_ + s;
#pragma unroll
    for (int d = 0; d < HD_; ++d) out[(size_t)d * S_] = in[d];
}

// ---------------------------------------------------------------------------
// Sliding-window causal flash attention, 32 query rows per wave.
// ---------------------------------------------------------------------------
__global__ __launch_bounds__(32) void k_attn(const f16* __restrict__ q,
                                             const f16* __restrict__ k,
                                             const f16* __restrict__ vt,
                                             f16* __restrict__ o) {
    int tile = blockIdx.x;
    int qt = tile % (S_ / 32);
    int h  = (tile / (S_ / 32)) % H_;
    int b  = tile / ((S_ / 32) * H_);
    int hk = h / NREP_;
    int qs = qt * 32;
    int lane = threadIdx.x;
    int m = lane & 15, hi = lane >> 4;

    __shared__ __align__(16) float sS[32 * 32];
    __shared__ __align__(16) f16   sP[32 * 32];
    __shared__ float sRow[32];

    const f16* qb = q  + ((size_t)(b * H_  + h ) * S_ + qs) * HD_;
    const f16* kb = k  + ((size_t)(b * HK_ + hk) * S_) * HD_;
    const f16* vb = vt + ((size_t)(b * HK_ + hk) * HD_) * S_;

    v16h aq00 = load_frag(qb, HD_, 0);
    v16h aq01 = load_frag(qb, HD_, 32);
    v16h aq10 = load_frag(qb + (size_t)16 * HD_, HD_, 0);
    v16h aq11 = load_frag(qb + (size_t)16 * HD_, HD_, 32);

    v8f acc[2][4] = {};
    float rm = -1e9f, rs = 0.0f;

    int lo = qs - (WINDOW_ - 1);
    if (lo < 0) lo = 0;
    lo &= ~31;

    for (int j0 = lo; j0 <= qs + 31; j0 += 32) {
        // ---- issue all fragment loads for this chunk up-front ----
        const f16* kr = kb + (size_t)j0 * HD_;
        v16h bk00 = load_frag(kr, HD_, 0);
        v16h bk01 = load_frag(kr, HD_, 32);
        v16h bk10 = load_frag(kr + (size_t)16 * HD_, HD_, 0);
        v16h bk11 = load_frag(kr + (size_t)16 * HD_, HD_, 32);
        v16h bv[4];
        bv[0] = load_frag(vb + (size_t)0  * S_, S_, j0);
        bv[1] = load_frag(vb + (size_t)16 * S_, S_, j0);
        bv[2] = load_frag(vb + (size_t)32 * S_, S_, j0);
        bv[3] = load_frag(vb + (size_t)48 * S_, S_, j0);

        int jn = j0 + 32;                       // prefetch next chunk
        if (jn <= qs + 31) {
            __builtin_prefetch(kb + (size_t)(jn + m) * HD_ + hi * 8, 0, 3);
            __builtin_prefetch(kb + (size_t)(jn + 16 + m) * HD_ + hi * 8, 0, 3);
            __builtin_prefetch(vb + (size_t)m * S_ + jn + hi * 8, 0, 3);
            __builtin_prefetch(vb + (size_t)(32 + m) * S_ + jn + hi * 8, 0, 3);
        }

        // ---- scores S[32 x 32] ----
        v8f c00 = {}, c01 = {}, c10 = {}, c11 = {};
        c00 = __builtin_amdgcn_wmma_f32_16x16x32_f16(false, aq00, false, bk00, (short)0, c00, false, false);
        c00 = __builtin_amdgcn_wmma_f32_16x16x32_f16(false, aq01, false, bk01, (short)0, c00, false, false);
        c01 = __builtin_amdgcn_wmma_f32_16x16x32_f16(false, aq00, false, bk10, (short)0, c01, false, false);
        c01 = __builtin_amdgcn_wmma_f32_16x16x32_f16(false, aq01, false, bk11, (short)0, c01, false, false);
        c10 = __builtin_amdgcn_wmma_f32_16x16x32_f16(false, aq10, false, bk00, (short)0, c10, false, false);
        c10 = __builtin_amdgcn_wmma_f32_16x16x32_f16(false, aq11, false, bk01, (short)0, c10, false, false);
        c11 = __builtin_amdgcn_wmma_f32_16x16x32_f16(false, aq10, false, bk10, (short)0, c11, false, false);
        c11 = __builtin_amdgcn_wmma_f32_16x16x32_f16(false, aq11, false, bk11, (short)0, c11, false, false);
#pragma unroll
        for (int r = 0; r < 8; ++r) {
            sS[(r + 8 * hi) * 32 + m]           = c00[r];
            sS[(r + 8 * hi) * 32 + 16 + m]      = c01[r];
            sS[(16 + r + 8 * hi) * 32 + m]      = c10[r];
            sS[(16 + r + 8 * hi) * 32 + 16 + m] = c11[r];
        }
        __syncthreads();

        // ---- masked online softmax: lane t owns query row qs+t (no divergence)
        {
            int i = qs + lane;
            float sv[32];
            float mx = rm;
#pragma unroll
            for (int jj = 0; jj < 32; ++jj) {
                int j = j0 + jj;
                float x = sS[lane * 32 + jj];
                bool ok = (j <= i) && (i - j < WINDOW_);
                sv[jj] = ok ? x : -1e9f;
                mx = fmaxf(mx, sv[jj]);
            }
            float alpha = __expf(rm - mx);
            float ssum = 0.0f;
#pragma unroll
            for (int jj = 0; jj < 32; ++jj) {
                float p = __expf(sv[jj] - mx);
                ssum += p;
                sP[lane * 32 + jj] = (f16)p;
            }
            rs = rs * alpha + ssum;
            rm = mx;
            sRow[lane] = alpha;
        }
        __syncthreads();

        // ---- rescale accumulators, add P(32x32) . V(32x64) ----
        float ar0[8], ar1[8];
#pragma unroll
        for (int r = 0; r < 8; ++r) {
            ar0[r] = sRow[r + 8 * hi];
            ar1[r] = sRow[16 + r + 8 * hi];
        }
        v16h ap0 = load_frag_lds(sP, 32, 0);
        v16h ap1 = load_frag_lds(sP + 16 * 32, 32, 0);
#pragma unroll
        for (int t = 0; t < 4; ++t) {
#pragma unroll
            for (int r = 0; r < 8; ++r) {
                acc[0][t][r] *= ar0[r];
                acc[1][t][r] *= ar1[r];
            }
            acc[0][t] = __builtin_amdgcn_wmma_f32_16x16x32_f16(false, ap0, false, bv[t], (short)0, acc[0][t], false, false);
            acc[1][t] = __builtin_amdgcn_wmma_f32_16x16x32_f16(false, ap1, false, bv[t], (short)0, acc[1][t], false, false);
        }
        __syncthreads();
    }

    // ---- epilogue: divide by row sums, store fp16 ----
    sRow[lane] = 1.0f / rs;
    __syncthreads();
    float ir0[8], ir1[8];
#pragma unroll
    for (int r = 0; r < 8; ++r) {
        ir0[r] = sRow[r + 8 * hi];
        ir1[r] = sRow[16 + r + 8 * hi];
    }
#pragma unroll
    for (int mt = 0; mt < 2; ++mt) {
#pragma unroll
        for (int t = 0; t < 4; ++t) {
#pragma unroll
            for (int r = 0; r < 8; ++r) {
                size_t row = (size_t)(b * S_ + qs + mt * 16 + r + 8 * hi);
                size_t col = (size_t)(h * HD_ + t * 16 + m);
                float iv = mt ? ir1[r] : ir0[r];
                o[row * (H_ * HD_) + col] = (f16)(acc[mt][t][r] * iv);
            }
        }
    }
}

// ---------------------------------------------------------------------------
extern "C" void kernel_launch(void* const* d_in, const int* in_sizes, int n_in,
                              void* d_out, int out_size, void* d_ws, size_t ws_size,
                              hipStream_t stream) {
    const float* hs = (const float*)d_in[0];
    // d_in[1] attention_mask, d_in[2] position_ids: recomputed analytically
    const float* Wq = (const float*)d_in[3];
    const float* Wk = (const float*)d_in[4];
    const float* Wv = (const float*)d_in[5];
    const float* Wo = (const float*)d_in[6];
    float* out = (float*)d_out;

    char* ws = (char*)d_ws;
    size_t off = 0;
    auto alloc = [&](size_t bytes) {
        char* p = ws + off;
        off = (off + bytes + 255) & ~(size_t)255;
        return p;
    };
    f16* Xh   = (f16*)alloc((size_t)M_ * D_ * 2);
    f16* Wqh  = (f16*)alloc((size_t)(H_ * HD_) * D_ * 2);
    f16* Wkh  = (f16*)alloc((size_t)(HK_ * HD_) * D_ * 2);
    f16* Wvh  = (f16*)alloc((size_t)(HK_ * HD_) * D_ * 2);
    f16* Woh  = (f16*)alloc((size_t)D_ * (H_ * HD_) * 2);
    f16* qpre = (f16*)alloc((size_t)M_ * (H_ * HD_) * 2);
    f16* kpre = (f16*)alloc((size_t)M_ * (HK_ * HD_) * 2);
    f16* vpre = (f16*)alloc((size_t)M_ * (HK_ * HD_) * 2);
    f16* qbuf = (f16*)alloc((size_t)B_ * H_ * S_ * HD_ * 2);
    f16* kbuf = (f16*)alloc((size_t)B_ * HK_ * S_ * HD_ * 2);
    f16* vtb  = (f16*)alloc((size_t)B_ * HK_ * HD_ * S_ * 2);
    f16* obuf = (f16*)alloc((size_t)M_ * (H_ * HD_) * 2);
    (void)ws_size; (void)in_sizes; (void)n_in; (void)out_size;

    // 1. fp32 -> fp16 conversions
    k_cvt<<<2048, 256, 0, stream>>>(hs, Xh, M_ * D_);
    k_cvt<<<1024, 256, 0, stream>>>(Wq, Wqh, H_ * HD_ * D_);
    k_cvt<<<256,  256, 0, stream>>>(Wk, Wkh, HK_ * HD_ * D_);
    k_cvt<<<256,  256, 0, stream>>>(Wv, Wvh, HK_ * HD_ * D_);
    k_cvt<<<1024, 256, 0, stream>>>(Wo, Woh, D_ * H_ * HD_);

    // 2. QKV projections (WMMA + async LDS W-tiles)
    {
        int wq = (M_ / 32) * ((H_ * HD_) / 64);   // 2048 waves
        int wk = (M_ / 32) * ((HK_ * HD_) / 64);  // 512 waves
        k_gemm<true><<<(wq + 7) / 8, 256, 0, stream>>>(Xh, Wqh, qpre, M_, H_ * HD_, D_);
        k_gemm<true><<<(wk + 7) / 8, 256, 0, stream>>>(Xh, Wkh, kpre, M_, HK_ * HD_, D_);
        k_gemm<true><<<(wk + 7) / 8, 256, 0, stream>>>(Xh, Wvh, vpre, M_, HK_ * HD_, D_);
    }

    // 3. RoPE + layout packing
    k_rope_q<<<(B_ * S_ * H_ + 255) / 256, 256, 0, stream>>>(qpre, qbuf);
    k_rope_k<<<(B_ * S_ * HK_ + 255) / 256, 256, 0, stream>>>(kpre, kbuf);
    k_pack_v<<<(B_ * S_ * HK_ + 255) / 256, 256, 0, stream>>>(vpre, vtb);

    // 4. sliding-window flash attention (WMMA), one wave per 32-query tile
    k_attn<<<B_ * H_ * (S_ / 32), 32, 0, stream>>>(qbuf, kbuf, vtb, obuf);

    // 5. output projection -> fp32 d_out
    {
        int wo = (M_ / 32) * (D_ / 64);  // 2048 waves
        k_gemm<false><<<(wo + 7) / 8, 256, 0, stream>>>(obuf, Woh, out, M_, D_, D_);
    }
}
